// LinearAttention_45105746543201
// MI455X (gfx1250) — matrix operand order repaired and checked by reading, not compiled, not run on hardware
//
#include <hip/hip_runtime.h>

// Problem constants (from reference setup_inputs)
#define BD 4      // batch
#define FD 1024   // model channels
#define SD 2048   // sequence
#define ID 2048   // inner channels
#define KT 5      // conv taps

typedef unsigned short u16;
typedef __attribute__((ext_vector_type(16))) __bf16 v16bf;
typedef __attribute__((ext_vector_type(8)))  float  v8f;

// Scheduling-intent hints: pin groups of instructions in the scheduler.
// 0x100 = DS-read class, 0x008 = MFMA/WMMA class.
#if defined(__has_builtin)
#if __has_builtin(__builtin_amdgcn_sched_group_barrier)
#define SCHED_GROUP(mask, n, id) __builtin_amdgcn_sched_group_barrier(mask, n, id)
#endif
#endif
#ifndef SCHED_GROUP
#define SCHED_GROUP(mask, n, id)
#endif

__device__ __forceinline__ u16 f2bf(float f) {
    unsigned u = __float_as_uint(f);
    u += 0x7fffu + ((u >> 16) & 1u);   // round-to-nearest-even
    return (u16)(u >> 16);
}

// ---------------------------------------------------------------------------
// out = bf16(inp + pos_embd)
__global__ __launch_bounds__(256)
void prep_kernel(const float* __restrict__ inp, const float* __restrict__ pos,
                 u16* __restrict__ xb) {
    const int idx = blockIdx.x * 256 + threadIdx.x;          // < BD*FD*SD
    const int fs  = idx & (FD * SD - 1);                     // FD*SD is 2^21
    xb[idx] = f2bf(inp[idx] + pos[fs]);
}

// fp32 -> bf16 weight convert
__global__ __launch_bounds__(256)
void convert_kernel(const float* __restrict__ in, u16* __restrict__ out, int n) {
    const int idx = blockIdx.x * 256 + threadIdx.x;
    if (idx < n) out[idx] = f2bf(in[idx]);
}

// ---------------------------------------------------------------------------
// C[M,N](f32) = A[M,K](bf16, lda) @ B[K,N](bf16, ldb), batched over blockIdx.z
// 128x128 block tile, 8 waves, wave = 64x32 = 4x2 WMMA 16x16x32 bf16 tiles.
// Double-buffered LDS, padded rows against bank conflicts. sched_group_barrier
// pins the schedule to: 12 DS reads (of 8 A b128 + 32 B u16 fragment loads) in flight, then the 8
// WMMAs back-to-back on the XDL pipe (partial dscnt waits instead of drains).
__global__ __launch_bounds__(256)
void gemm_bf16_wmma(float* __restrict__ C, const u16* __restrict__ A,
                    const u16* __restrict__ Bm,
                    int K, int lda, int ldb, int ldc,
                    long strideB, long strideC) {
    constexpr int AS = 40;    // 32 + 8 pad (bf16 elems per A-tile row)
    constexpr int BS = 136;   // 128 + 8 pad (bf16 elems per B-tile row)
    __shared__ u16 sA[2][128 * AS];
    __shared__ u16 sB[2][32 * BS];

    const int bn = blockIdx.x * 128;
    const int bm = blockIdx.y * 128;
    Bm += (long)blockIdx.z * strideB;
    C  += (long)blockIdx.z * strideC;

    const int tid   = threadIdx.x;
    const int lane  = tid & 31;        // wave32
    const int wid   = tid >> 5;        // 8 waves
    const int waveM = (wid & 1) * 64;
    const int waveN = (wid >> 1) * 32;
    const int half  = lane >> 4;
    const int mloc  = lane & 15;

    // global->LDS staging: 512 uint4 per tile, 2 per thread
    const int arow = tid >> 2, aseg = tid & 3;    // A: 128 rows x 4 segs
    const int brow = tid >> 4, bseg = tid & 15;   // B: 32 rows x 16 segs
    const u16* Ag = A  + (long)(bm + arow) * lda + aseg * 8;
    const u16* Bg = Bm + (long)brow * ldb + bn + bseg * 8;
    const long AgStep = (long)64 * lda;
    const long BgStep = (long)16 * ldb;

    const int saOff0 = arow * AS + aseg * 8;
    const int saOff1 = saOff0 + 64 * AS;
    const int sbOff0 = brow * BS + bseg * 8;
    const int sbOff1 = sbOff0 + 16 * BS;

    // per-lane WMMA fragment addresses (ISA 7.12.2 layouts)
    const int faBase = (waveM + mloc) * AS + half * 8;  // A: row=M, halves of K
    const int fbBase = half * 8 * BS + waveN + mloc;    // B: col=N, halves of K

    const v8f vzero = {0.f,0.f,0.f,0.f,0.f,0.f,0.f,0.f};
    v8f acc[4][2];
#pragma unroll
    for (int i = 0; i < 4; ++i)
#pragma unroll
        for (int j = 0; j < 2; ++j) acc[i][j] = vzero;

    const int nk = K >> 5;

    uint4 ra0 = *(const uint4*)(Ag);
    uint4 ra1 = *(const uint4*)(Ag + AgStep);
    uint4 rb0 = *(const uint4*)(Bg);
    uint4 rb1 = *(const uint4*)(Bg + BgStep);
    *(uint4*)&sA[0][saOff0] = ra0;
    *(uint4*)&sA[0][saOff1] = ra1;
    *(uint4*)&sB[0][sbOff0] = rb0;
    *(uint4*)&sB[0][sbOff1] = rb1;
    __syncthreads();

    union Frag { uint4 q[2]; v16bf v; };

    for (int kt = 0; kt < nk; ++kt) {
        const int  cur  = kt & 1;
        const bool more = (kt + 1) < nk;
        if (more) {   // prefetch next K-slab into registers
            const u16* Ap = Ag + (kt + 1) * 32;
            const u16* Bp = Bg + (long)(kt + 1) * 32 * ldb;
            ra0 = *(const uint4*)(Ap);
            ra1 = *(const uint4*)(Ap + AgStep);
            rb0 = *(const uint4*)(Bp);
            rb1 = *(const uint4*)(Bp + BgStep);
        }

        const u16* sa = sA[cur];
        const u16* sb = sB[cur];

        // All fragment loads into distinct registers first, ...
        Frag bf[2], af[4];
#pragma unroll
        for (int ni = 0; ni < 2; ++ni) {
            u16* be = (u16*)&bf[ni];
#pragma unroll
            for (int s = 0; s < 16; ++s)
                be[s] = sb[fbBase + ni * 16 + (s < 8 ? s : s + 8) * BS];
        }
#pragma unroll
        for (int mi = 0; mi < 4; ++mi) {
            af[mi].q[0] = *(const uint4*)&sa[faBase + mi * 16 * AS];
            af[mi].q[1] = *(const uint4*)&sa[faBase + mi * 16 * AS + 16];
        }
        // ... then the 8 WMMAs back-to-back.
#pragma unroll
        for (int mi = 0; mi < 4; ++mi)
#pragma unroll
            for (int ni = 0; ni < 2; ++ni)
                acc[mi][ni] = __builtin_amdgcn_wmma_f32_16x16x32_bf16(
                    false, af[mi].v, false, bf[ni].v, (short)0, acc[mi][ni],
                    false, false);

        // Pin the schedule for this region: 12 DS reads, then 8 WMMAs.
        SCHED_GROUP(0x100, 12, 0);
        SCHED_GROUP(0x008,  8, 0);

        if (more) {
            const int nxt = cur ^ 1;
            *(uint4*)&sA[nxt][saOff0] = ra0;
            *(uint4*)&sA[nxt][saOff1] = ra1;
            *(uint4*)&sB[nxt][sbOff0] = rb0;
            *(uint4*)&sB[nxt][sbOff1] = rb1;
            __syncthreads();
        }
    }

    // C/D layout: lane 0-15 -> N=lane, M=r; lane 16-31 -> M=r+8
#pragma unroll
    for (int mi = 0; mi < 4; ++mi)
#pragma unroll
        for (int ni = 0; ni < 2; ++ni) {
            const int col  = bn + waveN + ni * 16 + mloc;
            const int row0 = bm + waveM + mi * 16 + half * 8;
#pragma unroll
            for (int r = 0; r < 8; ++r)
                C[(long)(row0 + r) * ldc + col] = acc[mi][ni][r];
        }
}

// ---------------------------------------------------------------------------
// activate_norm over S of H[b, c, :] (fp32), scatter bf16 result into im2col
// buffer B2: B2[(c*5+t), n] = H1n[c, n+t-4], zero causal edge.
__global__ __launch_bounds__(256)
void norm_im2col_kernel(u16* __restrict__ B2, const float* __restrict__ H) {
    const int c = blockIdx.x, b = blockIdx.y;
    const float* row = H + ((long)b * ID + c) * SD;
    __shared__ float rs[256], rss[256];
    float vals[8], s = 0.f, ss = 0.f;
#pragma unroll
    for (int i = 0; i < 8; ++i) {
        float v = fmaxf(row[threadIdx.x + i * 256], 0.f);
        vals[i] = v; s += v; ss += v * v;
    }
    rs[threadIdx.x] = s; rss[threadIdx.x] = ss;
    __syncthreads();
    for (int off = 128; off > 0; off >>= 1) {
        if (threadIdx.x < off) {
            rs[threadIdx.x]  += rs[threadIdx.x + off];
            rss[threadIdx.x] += rss[threadIdx.x + off];
        }
        __syncthreads();
    }
    const float m   = rs[0] * (1.f / SD);
    const float ssc = fmaxf(rss[0] - (float)SD * m * m, 0.f);
    const float scl = sqrtf((float)SD) / (sqrtf(ssc) + 1e-5f);

    u16* out = B2 + ((long)b * (ID * KT) + (long)c * KT) * SD;
#pragma unroll
    for (int i = 0; i < 8; ++i) {
        const int j  = threadIdx.x + i * 256;
        const u16 bv = f2bf((vals[i] - m) * scl);
#pragma unroll
        for (int t = 0; t < KT; ++t) {
            const int n = j + 4 - t;
            if (n < SD) out[(long)t * SD + n] = bv;
        }
    }
    if (threadIdx.x < 4) {
        const int j = threadIdx.x;
#pragma unroll
        for (int t = 0; t < KT; ++t)
            if (j < 4 - t) out[(long)t * SD + j] = 0;
    }
}

// activate_norm over S of H[b, c, :] (fp32) -> bf16, plain layout
__global__ __launch_bounds__(256)
void norm_bf16_kernel(u16* __restrict__ O, const float* __restrict__ H) {
    const int c = blockIdx.x, b = blockIdx.y;
    const long base = ((long)b * ID + c) * SD;
    const float* row = H + base;
    __shared__ float rs[256], rss[256];
    float vals[8], s = 0.f, ss = 0.f;
#pragma unroll
    for (int i = 0; i < 8; ++i) {
        float v = fmaxf(row[threadIdx.x + i * 256], 0.f);
        vals[i] = v; s += v; ss += v * v;
    }
    rs[threadIdx.x] = s; rss[threadIdx.x] = ss;
    __syncthreads();
    for (int off = 128; off > 0; off >>= 1) {
        if (threadIdx.x < off) {
            rs[threadIdx.x]  += rs[threadIdx.x + off];
            rss[threadIdx.x] += rss[threadIdx.x + off];
        }
        __syncthreads();
    }
    const float m   = rs[0] * (1.f / SD);
    const float ssc = fmaxf(rss[0] - (float)SD * m * m, 0.f);
    const float scl = sqrtf((float)SD) / (sqrtf(ssc) + 1e-5f);
#pragma unroll
    for (int i = 0; i < 8; ++i) {
        const int j = threadIdx.x + i * 256;
        O[base + j] = f2bf((vals[i] - m) * scl);
    }
}

// cumsum over channels per (b,s) column; tmp = cum/div*scale + shift
__global__ __launch_bounds__(256)
void combine_kernel(float* __restrict__ tmp, const float* __restrict__ dpt,
                    const float* __restrict__ scl, const float* __restrict__ sft,
                    const float* __restrict__ dvs) {
    const int idx = blockIdx.x * 256 + threadIdx.x;   // < BD*SD
    const int b  = idx / SD;
    const int s0 = idx - b * SD;
    const long base = (long)b * FD * SD + s0;
    float c = 0.f;
    for (int f = 0; f < FD; ++f) {
        const long o = base + (long)f * SD;
        c += dpt[o];
        tmp[o] = c / dvs[(long)f * SD + s0] * scl[o] + sft[o];
    }
}

// out = inp + activate_norm(tmp) over S per (b,f) row (fp32)
__global__ __launch_bounds__(256)
void norm_residual_kernel(float* __restrict__ out, const float* __restrict__ tmp,
                          const float* __restrict__ inp) {
    const int f = blockIdx.x, b = blockIdx.y;
    const long base = ((long)b * FD + f) * SD;
    __shared__ float rs[256], rss[256];
    float vals[8], s = 0.f, ss = 0.f;
#pragma unroll
    for (int i = 0; i < 8; ++i) {
        float v = fmaxf(tmp[base + threadIdx.x + i * 256], 0.f);
        vals[i] = v; s += v; ss += v * v;
    }
    rs[threadIdx.x] = s; rss[threadIdx.x] = ss;
    __syncthreads();
    for (int off = 128; off > 0; off >>= 1) {
        if (threadIdx.x < off) {
            rs[threadIdx.x]  += rs[threadIdx.x + off];
            rss[threadIdx.x] += rss[threadIdx.x + off];
        }
        __syncthreads();
    }
    const float m   = rs[0] * (1.f / SD);
    const float ssc = fmaxf(rss[0] - (float)SD * m * m, 0.f);
    const float scl = sqrtf((float)SD) / (sqrtf(ssc) + 1e-5f);
#pragma unroll
    for (int i = 0; i < 8; ++i) {
        const int j = threadIdx.x + i * 256;
        out[base + j] = inp[base + j] + (vals[i] - m) * scl;
    }
}

// ---------------------------------------------------------------------------
extern "C" void kernel_launch(void* const* d_in, const int* in_sizes, int n_in,
                              void* d_out, int out_size, void* d_ws, size_t ws_size,
                              hipStream_t stream) {
    (void)in_sizes; (void)n_in; (void)out_size; (void)ws_size;
    const float* inp = (const float*)d_in[0];
    const float* pos = (const float*)d_in[1];
    const float* dvs = (const float*)d_in[2];
    const float* w0s[3] = {(const float*)d_in[3], (const float*)d_in[6], (const float*)d_in[9]};
    const float* w1s[3] = {(const float*)d_in[4], (const float*)d_in[7], (const float*)d_in[10]};
    const float* w2s[3] = {(const float*)d_in[5], (const float*)d_in[8], (const float*)d_in[11]};

    char* p = (char*)d_ws;
    auto alloc = [&](size_t bytes) {
        char* r = p; p += (bytes + 255) & ~(size_t)255; return r;
    };
    u16*   xb   = (u16*)  alloc((size_t)BD * FD * SD * 2);       // bf16(inp+pos)
    u16*   w0b  = (u16*)  alloc((size_t)ID * FD * 2);
    u16*   w1b  = (u16*)  alloc((size_t)ID * ID * KT * 2);
    u16*   w2b  = (u16*)  alloc((size_t)FD * ID * 2);
    float* Hbuf = (float*)alloc((size_t)BD * ID * SD * 4);       // H1 / H2 / tmp (serial reuse)
    u16*   B2   = (u16*)  alloc((size_t)BD * ID * KT * SD * 2);  // im2col bf16
    u16*   H2n  = (u16*)  alloc((size_t)BD * ID * SD * 2);
    float* outs[3];
    for (int i = 0; i < 3; ++i) outs[i] = (float*)alloc((size_t)BD * FD * SD * 4);

    const dim3 blk(256);
    prep_kernel<<<(BD * FD * SD) / 256, blk, 0, stream>>>(inp, pos, xb);

    for (int br = 0; br < 3; ++br) {
        convert_kernel<<<(ID * FD + 255) / 256, blk, 0, stream>>>(w0s[br], w0b, ID * FD);
        convert_kernel<<<(ID * ID * KT + 255) / 256, blk, 0, stream>>>(w1s[br], w1b, ID * ID * KT);
        convert_kernel<<<(FD * ID + 255) / 256, blk, 0, stream>>>(w2s[br], w2b, FD * ID);

        // H1[I,S] = W0[I,F] @ X[F,S]      (batched over z)
        gemm_bf16_wmma<<<dim3(SD / 128, ID / 128, BD), blk, 0, stream>>>(
            Hbuf, w0b, xb, FD, FD, SD, SD, (long)FD * SD, (long)ID * SD);
        norm_im2col_kernel<<<dim3(ID, BD), blk, 0, stream>>>(B2, Hbuf);
        // H2[I,S] = W1'[I,5I] @ B2[5I,S]  (causal K=5 conv as one GEMM, K=10240)
        gemm_bf16_wmma<<<dim3(SD / 128, ID / 128, BD), blk, 0, stream>>>(
            Hbuf, w1b, B2, ID * KT, ID * KT, SD, SD, (long)ID * KT * SD, (long)ID * SD);
        norm_bf16_kernel<<<dim3(ID, BD), blk, 0, stream>>>(H2n, Hbuf);
        // out[F,S] = W2[F,I] @ H2n[I,S]
        gemm_bf16_wmma<<<dim3(SD / 128, FD / 128, BD), blk, 0, stream>>>(
            outs[br], w2b, H2n, ID, ID, SD, SD, (long)ID * SD, (long)FD * SD);
    }

    combine_kernel<<<(BD * SD) / 256, blk, 0, stream>>>(Hbuf, outs[0], outs[1], outs[2], dvs);
    norm_residual_kernel<<<dim3(FD, BD), blk, 0, stream>>>((float*)d_out, Hbuf, inp);
}
